// TransformerLM_78967268704383
// MI455X (gfx1250) — matrix operand order repaired and checked by reading, not compiled, not run on hardware
//
#include <hip/hip_runtime.h>
#include <stdint.h>
#include <math.h>

#define DEVFN __device__ __forceinline__

// Enable CDNA5 async global->LDS copies (ASYNCcnt path) in the GEMM A-tile stage.
#define ATHENA_ASYNC_LDS 1

typedef _Float16 h16_t;
typedef __attribute__((ext_vector_type(16))) _Float16 v16h;
typedef __attribute__((ext_vector_type(8)))  _Float16 v8h;
typedef __attribute__((ext_vector_type(8)))  float    v8f;

constexpr int TV   = 32000;
constexpr int TD   = 512;
constexpr int TH   = 8;
constexpr int TDFF = 2048;
constexpr int TL   = 4;
constexpr int TT   = 2048;
constexpr int TB   = 2;
constexpr int BT   = TB * TT;   // 4096 activation rows
constexpr int D3   = 3 * TD;    // 1536

// ---------------------------------------------------------------- WMMA core
DEVFN v8f wmma32(v16h a, v16h b, v8f c) {
  return __builtin_amdgcn_wmma_f32_16x16x32_f16(false, a, false, b, (short)0, c,
                                                false, false);
}

// A-fragment: 16x32 f16, row-major with leading dim lda.
// lanes 0-15 -> rows, K = 0..15; lanes 16-31 -> K = 16..31.
DEVFN v16h load_afrag(const h16_t* base, int lda, int lane) {
  const int row = lane & 15;
  const int kb  = (lane >> 4) * 16;
  const h16_t* p = base + (size_t)row * lda + kb;
  v8h lo = *(const v8h*)(p);
  v8h hi = *(const v8h*)(p + 8);
  v16h r;
#pragma unroll
  for (int i = 0; i < 8; ++i) { r[i] = lo[i]; r[i + 8] = hi[i]; }
  return r;
}

// B-fragment: 32x16 f16 given as Bt row-major [N=16, K=32], leading dim ldb.
// lanes 0-15 -> cols with K=0..15; lanes 16-31 -> K=16..31.
DEVFN v16h load_bfrag(const h16_t* base, int ldb, int lane) {
  const int col = lane & 15;
  const int kb  = (lane >> 4) * 16;
  const h16_t* p = base + (size_t)col * ldb + kb;
  v8h lo = *(const v8h*)(p);
  v8h hi = *(const v8h*)(p + 8);
  v16h r;
#pragma unroll
  for (int i = 0; i < 8; ++i) { r[i] = lo[i]; r[i + 8] = hi[i]; }
  return r;
}

// ------------------------------------------------------------ generic GEMM
// C[M,N] = A[M,K](f16) * Bt[N,K](f16)^T + bias, optional ReLU.
// Block: 256 threads = 8 waves as 2(M) x 4(N); wave tile 32x64 (2x4 frags,
// 8 WMMA / K-step). Block tile 64x256. A block-tile (64x32 = 4KB) is staged
// through LDS once per K-step via async global->LDS copies; B streams from
// global (L2-resident weights) with next-step prefetch.
template <bool W16, bool W32, bool RELU, bool NTSTORE>
__global__ __launch_bounds__(256) void gemm_kernel(
    const h16_t* __restrict__ A, const h16_t* __restrict__ Bt,
    const float* __restrict__ bias, h16_t* __restrict__ C16,
    float* __restrict__ C32, int M, int N, int K) {
  __shared__ __align__(16) h16_t Atile[64 * 32];
  const int tid  = threadIdx.x;
  const int lane = tid & 31;
  const int wave = tid >> 5;
  const int wm = wave >> 2, wn = wave & 3;
  const int mblk = blockIdx.y * 64;
  const int n0 = blockIdx.x * 256 + wn * 64;

  // staging assignment: 256 threads x 16B = 4KB tile
  const int srow = tid >> 2;            // 0..63
  const int sseg = (tid & 3) * 8;       // halves 0,8,16,24
#if defined(ATHENA_ASYNC_LDS)
  const uint32_t ldsOff =
      (uint32_t)(uintptr_t)(&Atile[0]) + (uint32_t)(srow * 32 + sseg) * 2u;
#endif

  v8f acc[2][4] = {};
  for (int k0 = 0; k0 < K; k0 += 32) {
    const h16_t* ga = A + (size_t)(mblk + srow) * K + k0 + sseg;
#if defined(ATHENA_ASYNC_LDS)
    asm volatile("global_load_async_to_lds_b128 %0, %1, off"
                 :: "v"(ldsOff), "v"((uint64_t)(uintptr_t)ga)
                 : "memory");
    asm volatile("s_wait_asynccnt 0" ::: "memory");
#else
    *(v8h*)&Atile[srow * 32 + sseg] = *(const v8h*)ga;
#endif
    __syncthreads();

    v16h a0 = load_afrag(&Atile[(wm * 32 + 0)  * 32], 32, lane);
    v16h a1 = load_afrag(&Atile[(wm * 32 + 16) * 32], 32, lane);
#pragma unroll
    for (int t = 0; t < 4; ++t) {
      const h16_t* bp = Bt + (size_t)(n0 + 16 * t) * K + k0;
      v16h b = load_bfrag(bp, K, lane);
      if (k0 + 32 < K) {
        const h16_t* pf = bp + (size_t)(lane & 15) * K + ((lane >> 4) * 16) + 32;
        __builtin_prefetch(pf, 0, 1);   // next K-step B lines -> global_prefetch_b8
      }
      acc[0][t] = wmma32(a0, b, acc[0][t]);
      acc[1][t] = wmma32(a1, b, acc[1][t]);
    }
    __syncthreads();   // protect Atile against next iteration's overwrite
  }

  const int rbase = (lane >> 4) * 8;
  const int nidx  = lane & 15;
#pragma unroll
  for (int f = 0; f < 2; ++f) {
    const int m0f = mblk + wm * 32 + f * 16;
#pragma unroll
    for (int t = 0; t < 4; ++t) {
      const int col = n0 + 16 * t + nidx;
      const float bv = bias[col];
#pragma unroll
      for (int r = 0; r < 8; ++r) {
        float v = acc[f][t][r] + bv;
        if (RELU) v = v > 0.f ? v : 0.f;
        const size_t idx = (size_t)(m0f + rbase + r) * N + col;
        if (W32) {
          if (NTSTORE) __builtin_nontemporal_store(v, &C32[idx]);
          else C32[idx] = v;
        }
        if (W16) C16[idx] = (h16_t)v;
      }
    }
  }
}

// ------------------------------------------------- embedding + positional enc
__global__ __launch_bounds__(256) void embed_kernel(
    const int* __restrict__ x, const float* __restrict__ tok_emb,
    float* __restrict__ h32, h16_t* __restrict__ h16) {
  const int bt  = blockIdx.x;
  const int tok = x[bt];
  const int t   = bt % TT;
  const float sq = 22.627416997969522f;  // sqrt(512)
#pragma unroll
  for (int e = threadIdx.x; e < TD; e += 256) {
    const int i = e >> 1;
    const float freq = __expf((float)(2 * i) * (-9.210340371976184f / 512.0f));
    const float ang  = (float)t * freq;
    const float pe   = (e & 1) ? __cosf(ang) : __sinf(ang);
    const float v    = tok_emb[(size_t)tok * TD + e] * sq + pe;
    h32[(size_t)bt * TD + e] = v;
    h16[(size_t)bt * TD + e] = (h16_t)v;
  }
}

// --------------------------------------------------- fused residual + LayerNorm
__global__ __launch_bounds__(256) void add_ln_kernel(
    const float* __restrict__ x, const float* __restrict__ dlt,
    const float* __restrict__ g, const float* __restrict__ bta,
    float* __restrict__ out32, h16_t* __restrict__ out16) {
  const int row  = blockIdx.x;
  const int tid  = threadIdx.x;
  const int lane = tid & 31, wv = tid >> 5;
  __shared__ float sred[8], sred2[8];
  const size_t off = (size_t)row * TD;

  float v0 = x[off + tid]       + dlt[off + tid];
  float v1 = x[off + tid + 256] + dlt[off + tid + 256];

  float s = v0 + v1;
#pragma unroll
  for (int o = 16; o >= 1; o >>= 1) s += __shfl_xor(s, o, 32);
  if (lane == 0) sred[wv] = s;
  __syncthreads();
  float tot = 0.f;
#pragma unroll
  for (int i = 0; i < 8; ++i) tot += sred[i];
  const float mean = tot * (1.0f / TD);

  const float d0 = v0 - mean, d1 = v1 - mean;
  float s2 = d0 * d0 + d1 * d1;
#pragma unroll
  for (int o = 16; o >= 1; o >>= 1) s2 += __shfl_xor(s2, o, 32);
  if (lane == 0) sred2[wv] = s2;
  __syncthreads();
  float tot2 = 0.f;
#pragma unroll
  for (int i = 0; i < 8; ++i) tot2 += sred2[i];
  const float rstd = rsqrtf(tot2 * (1.0f / TD) + 1e-5f);

  const float o0 = d0 * rstd * g[tid]       + bta[tid];
  const float o1 = d1 * rstd * g[tid + 256] + bta[tid + 256];
  out32[off + tid]       = o0;
  out32[off + tid + 256] = o1;
  out16[off + tid]       = (h16_t)o0;
  out16[off + tid + 256] = (h16_t)o1;
}

// ------------------------------------------- V transpose: [b,t,(v)h,dk] -> [b,h,dk,t]
__global__ __launch_bounds__(256) void vtrans_kernel(
    const h16_t* __restrict__ qkv16, h16_t* __restrict__ Vt) {
  const size_t idx = (size_t)blockIdx.x * 256 + threadIdx.x;
  const int t  = (int)(idx % TT);
  const int dk = (int)((idx / TT) & 63);
  const int hh = (int)((idx / (TT * 64)) & (TH - 1));
  const int bb = (int)(idx / ((size_t)TT * 64 * TH));
  Vt[idx] = qkv16[(size_t)(bb * TT + t) * D3 + 2 * TD + hh * 64 + dk];
}

// ------------------------------------------------ flash attention (1 wave/block)
// One 32-key block step of the online-softmax loop.  MASKED only for the
// (at most one) key block intersecting the causal diagonal.
template <bool MASKED>
DEVFN void attn_block(int kb, int q0, int lane, int rbase, int nidx,
                      const h16_t* __restrict__ kbase,
                      const h16_t* __restrict__ vbase,
                      h16_t* P, const v16h (&qa)[2],
                      float (&mrun)[8], float (&lrun)[8], v8f (&O)[4]) {
  float S[2][8];
#pragma unroll
  for (int t = 0; t < 2; ++t) {
    v8f acc = {};
#pragma unroll
    for (int s = 0; s < 2; ++s) {
      v16h kf = load_bfrag(kbase + (size_t)(kb + 16 * t) * D3 + 32 * s, D3, lane);
      acc = wmma32(qa[s], kf, acc);
    }
#pragma unroll
    for (int r = 0; r < 8; ++r) {
      float v = acc[r] * 0.125f;  // 1/sqrt(DK)
      if (MASKED && (kb + 16 * t + nidx > q0 + rbase + r)) v = -1.0e9f;
      S[t][r] = v;
    }
  }
  float rowmax[8];
#pragma unroll
  for (int r = 0; r < 8; ++r) rowmax[r] = fmaxf(S[0][r], S[1][r]);
#pragma unroll
  for (int o = 8; o >= 1; o >>= 1)
#pragma unroll
    for (int r = 0; r < 8; ++r)
      rowmax[r] = fmaxf(rowmax[r], __shfl_xor(rowmax[r], o, 32));

  float alpha[8], rsum[8];
#pragma unroll
  for (int r = 0; r < 8; ++r) {
    const float mn = fmaxf(mrun[r], rowmax[r]);
    alpha[r] = __expf(mrun[r] - mn);
    mrun[r] = mn;
    rsum[r] = 0.f;
  }
  __syncthreads();
#pragma unroll
  for (int t = 0; t < 2; ++t)
#pragma unroll
    for (int r = 0; r < 8; ++r) {
      const float p = __expf(S[t][r] - mrun[r]);
      rsum[r] += p;
      P[(rbase + r) * 32 + 16 * t + nidx] = (h16_t)p;
    }
#pragma unroll
  for (int o = 8; o >= 1; o >>= 1)
#pragma unroll
    for (int r = 0; r < 8; ++r) rsum[r] += __shfl_xor(rsum[r], o, 32);
#pragma unroll
  for (int r = 0; r < 8; ++r) lrun[r] = lrun[r] * alpha[r] + rsum[r];
#pragma unroll
  for (int t4 = 0; t4 < 4; ++t4)
#pragma unroll
    for (int r = 0; r < 8; ++r) O[t4][r] *= alpha[r];
  __syncthreads();

  v16h pa = load_afrag((const h16_t*)P, 32, lane);
#pragma unroll
  for (int t4 = 0; t4 < 4; ++t4) {
    v16h vf = load_bfrag(vbase + (size_t)(16 * t4) * TT + kb, TT, lane);
    O[t4] = wmma32(pa, vf, O[t4]);
  }
}

// grid = (T/16 query tiles, H, B); block = 1 wave.
__global__ __launch_bounds__(32) void attn_kernel(
    const h16_t* __restrict__ qkv16, const h16_t* __restrict__ Vt,
    h16_t* __restrict__ o16) {
  __shared__ __align__(16) h16_t P[16 * 32];
  const int lane = threadIdx.x;
  const int qt = blockIdx.x, hh = blockIdx.y, bb = blockIdx.z;
  const int q0 = qt * 16;

  const h16_t* qbase = qkv16 + (size_t)bb * TT * D3 + hh * 64;
  const h16_t* kbase = qbase + TD;
  const h16_t* vbase = Vt + (size_t)(bb * TH + hh) * 64 * TT;

  v16h qa[2];
#pragma unroll
  for (int s = 0; s < 2; ++s)
    qa[s] = load_afrag(qbase + (size_t)q0 * D3 + 32 * s, D3, lane);

  const int rbase = (lane >> 4) * 8;
  const int nidx  = lane & 15;
  float mrun[8], lrun[8];
#pragma unroll
  for (int r = 0; r < 8; ++r) { mrun[r] = -3.0e38f; lrun[r] = 0.f; }
  v8f O[4] = {};

  const int qfull = q0 & ~31;  // key blocks strictly below the diagonal
  for (int kb = 0; kb < qfull; kb += 32)
    attn_block<false>(kb, q0, lane, rbase, nidx, kbase, vbase, P, qa, mrun, lrun, O);
  for (int kb = qfull; kb < q0 + 16; kb += 32)
    attn_block<true>(kb, q0, lane, rbase, nidx, kbase, vbase, P, qa, mrun, lrun, O);

#pragma unroll
  for (int t4 = 0; t4 < 4; ++t4) {
    const int col = hh * 64 + 16 * t4 + nidx;
#pragma unroll
    for (int r = 0; r < 8; ++r) {
      const int row = bb * TT + q0 + rbase + r;
      o16[(size_t)row * TD + col] = (h16_t)(O[t4][r] / lrun[r]);
    }
  }
}

// ------------------------------------------------------- fp32 -> f16 convert
__global__ __launch_bounds__(256) void cvt_kernel(const float* __restrict__ src,
                                                  h16_t* __restrict__ dst,
                                                  size_t n) {
  size_t i = (size_t)blockIdx.x * 256 + threadIdx.x;
  const size_t stride = (size_t)gridDim.x * 256;
  for (; i < n; i += stride) dst[i] = (h16_t)src[i];
}

// =========================================================== host launcher
extern "C" void kernel_launch(void* const* d_in, const int* in_sizes, int n_in,
                              void* d_out, int out_size, void* d_ws, size_t ws_size,
                              hipStream_t stream) {
  const int*   x       = (const int*)d_in[0];
  const float* tok_emb = (const float*)d_in[1];
  const float* qkv_w   = (const float*)d_in[2];
  const float* qkv_b   = (const float*)d_in[3];
  const float* ow      = (const float*)d_in[4];
  const float* ob      = (const float*)d_in[5];
  const float* ln1_g   = (const float*)d_in[6];
  const float* ln1_b   = (const float*)d_in[7];
  const float* w1      = (const float*)d_in[8];
  const float* b1      = (const float*)d_in[9];
  const float* w2      = (const float*)d_in[10];
  const float* b2      = (const float*)d_in[11];
  const float* ln2_g   = (const float*)d_in[12];
  const float* ln2_b   = (const float*)d_in[13];
  const float* fc_w    = (const float*)d_in[14];
  const float* fc_b    = (const float*)d_in[15];
  float* out = (float*)d_out;

  uint8_t* p = (uint8_t*)d_ws;
  auto take = [&](size_t bytes) -> void* {
    void* r = (void*)p;
    p += (bytes + 255) & ~(size_t)255;
    return r;
  };
  float* h32     = (float*)take((size_t)BT * TD * 4);
  h16_t* h16     = (h16_t*)take((size_t)BT * TD * 2);
  float* delta32 = (float*)take((size_t)BT * TD * 4);
  h16_t* qkv16   = (h16_t*)take((size_t)BT * D3 * 2);
  h16_t* Vt      = (h16_t*)take((size_t)TB * TH * 64 * TT * 2);
  h16_t* o16     = (h16_t*)take((size_t)BT * TD * 2);
  h16_t* ffn16   = (h16_t*)take((size_t)BT * TDFF * 2);
  h16_t* wq16    = (h16_t*)take((size_t)TL * D3 * TD * 2);
  h16_t* wo16    = (h16_t*)take((size_t)TL * TD * TD * 2);
  h16_t* w116    = (h16_t*)take((size_t)TL * TDFF * TD * 2);
  h16_t* w216    = (h16_t*)take((size_t)TL * TD * TDFF * 2);
  h16_t* wfc16   = (h16_t*)take((size_t)TV * TD * 2);

  cvt_kernel<<<4096, 256, 0, stream>>>(qkv_w, wq16, (size_t)TL * D3 * TD);
  cvt_kernel<<<2048, 256, 0, stream>>>(ow, wo16, (size_t)TL * TD * TD);
  cvt_kernel<<<4096, 256, 0, stream>>>(w1, w116, (size_t)TL * TDFF * TD);
  cvt_kernel<<<4096, 256, 0, stream>>>(w2, w216, (size_t)TL * TD * TDFF);
  cvt_kernel<<<8192, 256, 0, stream>>>(fc_w, wfc16, (size_t)TV * TD);

  embed_kernel<<<BT, 256, 0, stream>>>(x, tok_emb, h32, h16);

  for (int l = 0; l < TL; ++l) {
    gemm_kernel<true, false, false, false><<<dim3(D3 / 256, BT / 64), 256, 0, stream>>>(
        h16, wq16 + (size_t)l * D3 * TD, qkv_b + (size_t)l * D3, qkv16, nullptr,
        BT, D3, TD);
    vtrans_kernel<<<(TB * TH * 64 * TT) / 256, 256, 0, stream>>>(qkv16, Vt);
    attn_kernel<<<dim3(TT / 16, TH, TB), 32, 0, stream>>>(qkv16, Vt, o16);
    gemm_kernel<false, true, false, false><<<dim3(TD / 256, BT / 64), 256, 0, stream>>>(
        o16, wo16 + (size_t)l * TD * TD, ob + (size_t)l * TD, nullptr, delta32,
        BT, TD, TD);
    add_ln_kernel<<<BT, 256, 0, stream>>>(h32, delta32, ln1_g + (size_t)l * TD,
                                          ln1_b + (size_t)l * TD, h32, h16);
    gemm_kernel<true, false, true, false><<<dim3(TDFF / 256, BT / 64), 256, 0, stream>>>(
        h16, w116 + (size_t)l * TDFF * TD, b1 + (size_t)l * TDFF, ffn16, nullptr,
        BT, TDFF, TD);
    gemm_kernel<false, true, false, false><<<dim3(TD / 256, BT / 64), 256, 0, stream>>>(
        ffn16, w216 + (size_t)l * TD * TDFF, b2 + (size_t)l * TD, nullptr, delta32,
        BT, TD, TDFF);
    add_ln_kernel<<<BT, 256, 0, stream>>>(h32, delta32, ln2_g + (size_t)l * TD,
                                          ln2_b + (size_t)l * TD, h32, h16);
  }

  gemm_kernel<false, true, false, true><<<dim3(TV / 256, BT / 64), 256, 0, stream>>>(
      h16, wfc16, fc_b, nullptr, out, BT, TV, TD);
}
